// NF_35055523069990
// MI455X (gfx1250) — matrix operand order repaired and checked by reading, not compiled, not run on hardware
//
#include <hip/hip_runtime.h>
#include <hip/hip_bf16.h>
#include <math.h>

#define BTOT 131072
#define DDIM 64
#define HDIM 512
#define NLAY 3
#define NBLK 8
#define HALF 32
#define CHUNK 16384
#define EPSV 1e-5f

typedef __attribute__((ext_vector_type(16))) __bf16 v16bf;
typedef __attribute__((ext_vector_type(8)))  float  v8f;

#if __has_builtin(__builtin_amdgcn_global_load_async_to_lds_b128) && \
    __has_builtin(__builtin_amdgcn_s_wait_asynccnt)
#define NF_USE_ASYNC 1
#endif

typedef __attribute__((__vector_size__(16))) int v4i;
typedef __attribute__((address_space(1))) v4i gv4i;   // global 128-bit
typedef __attribute__((address_space(3))) v4i lv4i;   // LDS    128-bit

__device__ __forceinline__ unsigned short f2bf(float f) {
  unsigned int u = __float_as_uint(f);
  u += 0x7FFFu + ((u >> 16) & 1u);          // round-to-nearest-even
  return (unsigned short)(u >> 16);
}

// 16-byte global -> LDS copy (async gfx1250 path if available)
__device__ __forceinline__ void copy16_g2l(unsigned short* ldst,
                                           const unsigned short* gsrc) {
#ifdef NF_USE_ASYNC
  __builtin_amdgcn_global_load_async_to_lds_b128(
      (gv4i*)gsrc, (lv4i*)ldst, 0, 0);
#else
  *(uint4*)ldst = *(const uint4*)gsrc;
#endif
}
__device__ __forceinline__ void async_wait0() {
#ifdef NF_USE_ASYNC
  __builtin_amdgcn_s_wait_asynccnt(0);
#endif
}

// ---------------------------------------------------------------------------
// Weight convert fp32 -> bf16, pre-swizzled into B-fragment order:
//   dst[mat*K*N + ((k>>5)*(N/16) + (n>>4))*512 + lane*16 + e]
//   lane = (n&15) + 16*((k>>3)&1),  e = (k&7) | (((k>>4)&1)<<3)
// ---------------------------------------------------------------------------
__global__ __launch_bounds__(256)
void nf_cvtw(const float* __restrict__ src, unsigned short* __restrict__ dst,
             int K, int N, long total) {
  long i = (long)blockIdx.x * 256 + threadIdx.x;
  if (i >= total) return;
  long mn  = (long)K * N;
  long mat = i / mn;
  int  rem = (int)(i - mat * mn);
  int  k   = rem / N;
  int  n   = rem - k * N;
  int  ll  = (n & 15) + (((k >> 3) & 1) << 4);
  long off = ((long)(k >> 5) * (N >> 4) + (n >> 4)) * 512 + ll * 16 + ((k & 7) | (((k >> 4) & 1) << 3));
  dst[mat * mn + off] = f2bf(src[i]);
}

// ---------------------------------------------------------------------------
// bf16 WMMA GEMM:  Out[M,N] = act(A[M,K] @ W[K,N] + bias)
//   aIsF32=1 : A is fp32 row-major slice (lda, aOff)   [layer 0 only]
//   aIsF32=0 : A is bf16 in A-fragment order [(m>>4)*kSteps + ks][lane][e]
//   W        : bf16 pre-swizzled B-fragment order
//   mode 0: tanh->bf16(A-frag order), 1: relu->bf16(A-frag), 2: ident->f32 rowmajor
// Tile 128x64 per workgroup (8 waves, 32x32/wave), K-step 32.
// ---------------------------------------------------------------------------
__global__ __launch_bounds__(256)
void nf_gemm(const void* __restrict__ Aptr, int aIsF32, int lda, int aOff,
             const unsigned short* __restrict__ Wsw,
             const float* __restrict__ bias,
             void* __restrict__ Out, int mode,
             int M, int N, int K) {
  __shared__ __align__(32) unsigned short As[8 * 512];
  __shared__ __align__(32) unsigned short Bs[4 * 512];

  const int tid  = threadIdx.x;
  const int lane = tid & 31;
  const int w    = tid >> 5;
  const int wm   = w & 3;
  const int wn   = w >> 2;
  const int rowBase = blockIdx.x * 128;
  const int colBase = blockIdx.y * 64;
  const int nSubT   = N >> 4;                       // total 16-col subtiles
  const int cbSub   = colBase >> 4;
  const int nSub    = (nSubT - cbSub) < 4 ? (nSubT - cbSub) : 4;
  const int bUnits  = nSub * 64;                    // 16B units of B tile
  const int kSteps  = K >> 5;
  const bool waveActive = (colBase + wn * 32) < N;

  const float*          Af = (const float*)Aptr;
  const unsigned short* Ab = (const unsigned short*)Aptr;

  v8f c00 = {}, c01 = {}, c10 = {}, c11 = {};

  for (int ks = 0; ks < kSteps; ++ks) {
    if (aIsF32) {
      // layer-0: fp32 strided slice -> convert + swizzle (float4 wide)
#pragma unroll
      for (int i = 0; i < 2; ++i) {
        int chunk  = tid + 256 * i;                 // 0..511 : (row, kgroup)
        int mLocal = chunk >> 2;
        int g      = chunk & 3;
        const float4* src = (const float4*)
            &Af[(long)(rowBase + mLocal) * lda + aOff + ks * 32 + g * 8];
        float4 f0 = src[0], f1 = src[1];
        uint4 pk;
        pk.x = (unsigned)f2bf(f0.x) | ((unsigned)f2bf(f0.y) << 16);
        pk.y = (unsigned)f2bf(f0.z) | ((unsigned)f2bf(f0.w) << 16);
        pk.z = (unsigned)f2bf(f1.x) | ((unsigned)f2bf(f1.y) << 16);
        pk.w = (unsigned)f2bf(f1.z) | ((unsigned)f2bf(f1.w) << 16);
        int dst = (mLocal >> 4) * 512 +
                  ((mLocal & 15) + ((g & 1) << 4)) * 16 + ((g >> 1) << 3);
        *(uint4*)&As[dst] = pk;
      }
    } else {
      // hidden layers: activations already in fragment order -> 16B copies
#pragma unroll
      for (int i = 0; i < 2; ++i) {
        int unit = tid + 256 * i;                   // 0..511
        int m16i = unit >> 6;
        int off  = (unit & 63) * 8;
        const unsigned short* src =
            Ab + ((long)((rowBase >> 4) + m16i) * kSteps + ks) * 512 + off;
        copy16_g2l(&As[m16i * 512 + off], src);
      }
    }
    // B: pre-swizzled weights, contiguous 16B copies
    if (tid < bUnits) {
      const unsigned short* src =
          Wsw + ((long)ks * nSubT + cbSub) * 512 + tid * 8;
      copy16_g2l(&Bs[tid * 8], src);
    }
    async_wait0();
    __syncthreads();

    if (waveActive) {   // wave-uniform branch: EXEC all-ones for WMMA
      const v16bf* Asv = reinterpret_cast<const v16bf*>(As);
      const v16bf* Bsv = reinterpret_cast<const v16bf*>(Bs);
      v16bf fa0 = Asv[(wm * 2 + 0) * 32 + lane];
      v16bf fa1 = Asv[(wm * 2 + 1) * 32 + lane];
      v16bf fb0 = Bsv[(wn * 2 + 0) * 32 + lane];
      v16bf fb1 = Bsv[(wn * 2 + 1) * 32 + lane];
      c00 = __builtin_amdgcn_wmma_f32_16x16x32_bf16(false, fa0, false, fb0,
                                                    (short)0, c00, false, false);
      c01 = __builtin_amdgcn_wmma_f32_16x16x32_bf16(false, fa0, false, fb1,
                                                    (short)0, c01, false, false);
      c10 = __builtin_amdgcn_wmma_f32_16x16x32_bf16(false, fa1, false, fb0,
                                                    (short)0, c10, false, false);
      c11 = __builtin_amdgcn_wmma_f32_16x16x32_bf16(false, fa1, false, fb1,
                                                    (short)0, c11, false, false);
    }
    __syncthreads();
  }

  if (!waveActive) return;

  // C/D layout: lane 0-15 -> (M=j, N=lane); lane 16-31 -> (M=8+j, N=lane-16)
  const int mOff = (lane >> 4) * 8;
  const int nIdx = lane & 15;
  float*          Outf  = (float*)Out;
  unsigned short* Outb  = (unsigned short*)Out;
  const int oSteps = N >> 5;                        // consumer K-steps
#pragma unroll
  for (int sub = 0; sub < 4; ++sub) {
    v8f c = (sub == 0) ? c00 : (sub == 1) ? c01 : (sub == 2) ? c10 : c11;
    int mT  = rowBase + wm * 32 + ((sub >> 1) ? 16 : 0);
    int col = colBase + wn * 32 + ((sub & 1) ? 16 : 0) + nIdx;
    if (col >= N) continue;
    float bv = bias[col];
    if (mode == 2) {
#pragma unroll
      for (int j = 0; j < 8; ++j)
        Outf[(long)(mT + mOff + j) * N + col] = c[j] + bv;
    } else {
      int kk      = col & 31;
      int e2      = (kk & 7) | ((kk & 16) >> 1);
      int laneAdd = ((kk & 15) >> 3) << 4;
      long colB   = (long)(col >> 5) * 512 + e2;
#pragma unroll
      for (int j = 0; j < 8; ++j) {
        int  row = mT + mOff + j;
        float v  = c[j] + bv;
        v = (mode == 0) ? tanhf(v) : (v > 0.f ? v : 0.f);
        long idx = ((long)(row >> 4) * oSteps) * 512 + colB +
                   ((row & 15) + laneAdd) * 16;
        Outb[idx] = f2bf(v);
      }
    }
  }
}

// ---------------------------------------------------------------------------
// Coupling: zc = yc*exp(s)+t, assemble z, per-row sum(s) -> logdet,
// column sum/sumsq partials -> stats (wave32 = one row per wave).
// ---------------------------------------------------------------------------
__global__ __launch_bounds__(256)
void nf_couple(const float* __restrict__ sOut, const float* __restrict__ tOut,
               const float* __restrict__ yCur, float* __restrict__ z,
               float* __restrict__ logdet, float* __restrict__ stats,
               int rowBase, int parity) {
  __shared__ float sSum[64], sSq[64];
  int tid = threadIdx.x;
  if (tid < 64) { sSum[tid] = 0.f; sSq[tid] = 0.f; }
  __syncthreads();

  int lane = tid & 31;
  int rl   = blockIdx.x * 8 + (tid >> 5);
  long r   = rowBase + rl;
  int sCol = parity ? (32 + lane) : lane;
  int cCol = parity ? lane : (32 + lane);

  float s   = sOut[rl * 32 + lane];
  float t   = tOut[rl * 32 + lane];
  float ysv = yCur[r * 64 + sCol];
  float ycv = yCur[r * 64 + cCol];
  float zcv = ycv * expf(s) + t;

  z[r * 64 + sCol] = ysv;
  z[r * 64 + cCol] = zcv;

  float sum = s;
#pragma unroll
  for (int o = 16; o > 0; o >>= 1) sum += __shfl_down(sum, o, 32);
  if (lane == 0) logdet[r] += sum;

  atomicAdd(&sSum[sCol], ysv); atomicAdd(&sSq[sCol], ysv * ysv);
  atomicAdd(&sSum[cCol], zcv); atomicAdd(&sSq[cCol], zcv * zcv);
  __syncthreads();
  if (tid < 64) {
    atomicAdd(&stats[tid],      sSum[tid]);
    atomicAdd(&stats[64 + tid], sSq[tid]);
  }
}

// ---------------------------------------------------------------------------
__global__ void nf_bnstats(const float* __restrict__ stats,
                           const float* __restrict__ lg,
                           const float* __restrict__ beta,
                           float* __restrict__ bnp, int Btot) {
  __shared__ float red[64];
  int c = threadIdx.x;                      // 64 threads
  float sum = stats[c], sq = stats[64 + c];
  float mean = sum / (float)Btot;
  float var  = (sq - sum * mean) / (float)(Btot - 1);   // ddof=1
  bnp[c]       = mean;
  bnp[64 + c]  = rsqrtf(var + EPSV);
  bnp[128 + c] = expf(lg[c]);
  bnp[192 + c] = beta[c];
  red[c] = lg[c] - 0.5f * logf(var + EPSV);
  __syncthreads();
#pragma unroll
  for (int o = 32; o > 0; o >>= 1) {
    if (c < o) red[c] += red[c + o];
    __syncthreads();
  }
  if (c == 0) bnp[256] = red[0];
}

// ---------------------------------------------------------------------------
__global__ __launch_bounds__(256)
void nf_bnapply(const float* __restrict__ z, const float* __restrict__ bnp,
                float* __restrict__ yCur, float* __restrict__ logdet,
                long total) {
  long e = (long)blockIdx.x * 256 + threadIdx.x;
  if (e >= total) return;
  int c = (int)(e & 63);
  float xhat = (z[e] - bnp[c]) * bnp[64 + c];
  yCur[e] = bnp[128 + c] * xhat + bnp[192 + c];
  if (c == 0) logdet[e >> 6] += bnp[256];
}

// ---------------------------------------------------------------------------
extern "C" void kernel_launch(void* const* d_in, const int* in_sizes, int n_in,
                              void* d_out, int out_size, void* d_ws, size_t ws_size,
                              hipStream_t stream) {
  const float* y    = (const float*)d_in[0];
  const float* sW0  = (const float*)d_in[1];
  const float* sb0  = (const float*)d_in[2];
  const float* sWh  = (const float*)d_in[3];
  const float* sbh  = (const float*)d_in[4];
  const float* sWo  = (const float*)d_in[5];
  const float* sbo  = (const float*)d_in[6];
  const float* tW0  = (const float*)d_in[7];
  const float* tb0  = (const float*)d_in[8];
  const float* tWh  = (const float*)d_in[9];
  const float* tbh  = (const float*)d_in[10];
  const float* tWo  = (const float*)d_in[11];
  const float* tbo  = (const float*)d_in[12];
  const float* lg   = (const float*)d_in[13];
  const float* beta = (const float*)d_in[14];

  char* p = (char*)d_ws;
  auto take = [&](size_t bytes) -> char* {
    char* r = p; p += (bytes + 255) & ~(size_t)255; return r;
  };
  const size_t nW0 = (size_t)NBLK * HALF * HDIM;
  const size_t nWh = (size_t)NBLK * NLAY * HDIM * HDIM;
  const size_t nWo = (size_t)NBLK * HDIM * HALF;

  float* yCur   = (float*)take((size_t)BTOT * DDIM * 4);
  float* z      = (float*)take((size_t)BTOT * DDIM * 4);
  float* logdet = (float*)take((size_t)BTOT * 4);
  float* stats  = (float*)take(512);
  float* bnp    = (float*)take(2048);
  unsigned short* sW0b = (unsigned short*)take(nW0 * 2);
  unsigned short* sWhb = (unsigned short*)take(nWh * 2);
  unsigned short* sWob = (unsigned short*)take(nWo * 2);
  unsigned short* tW0b = (unsigned short*)take(nW0 * 2);
  unsigned short* tWhb = (unsigned short*)take(nWh * 2);
  unsigned short* tWob = (unsigned short*)take(nWo * 2);
  unsigned short* actS0 = (unsigned short*)take((size_t)CHUNK * HDIM * 2);
  unsigned short* actS1 = (unsigned short*)take((size_t)CHUNK * HDIM * 2);
  unsigned short* actT0 = (unsigned short*)take((size_t)CHUNK * HDIM * 2);
  unsigned short* actT1 = (unsigned short*)take((size_t)CHUNK * HDIM * 2);
  float* sout = (float*)take((size_t)CHUNK * HALF * 4);
  float* tout = (float*)take((size_t)CHUNK * HALF * 4);

  (void)hipMemcpyAsync(yCur, y, (size_t)BTOT * DDIM * 4,
                       hipMemcpyDeviceToDevice, stream);
  (void)hipMemsetAsync(logdet, 0, (size_t)BTOT * 4, stream);

  auto cvtw = [&](const float* src, unsigned short* dst, int K, int N, size_t n) {
    nf_cvtw<<<dim3((unsigned)((n + 255) / 256)), dim3(256), 0, stream>>>(
        src, dst, K, N, (long)n);
  };
  cvtw(sW0, sW0b, HALF, HDIM, nW0);
  cvtw(sWh, sWhb, HDIM, HDIM, nWh);
  cvtw(sWo, sWob, HDIM, HALF, nWo);
  cvtw(tW0, tW0b, HALF, HDIM, nW0);
  cvtw(tWh, tWhb, HDIM, HDIM, nWh);
  cvtw(tWo, tWob, HDIM, HALF, nWo);

  auto gemm = [&](const void* A, int aIsF32, int lda, int aOff,
                  const unsigned short* W, const float* bias,
                  void* out, int mode, int M, int N, int K) {
    dim3 g(M / 128, (N + 63) / 64);
    nf_gemm<<<g, 256, 0, stream>>>(A, aIsF32, lda, aOff, W, bias,
                                   out, mode, M, N, K);
  };

  for (int b = 0; b < NBLK; ++b) {
    (void)hipMemsetAsync(stats, 0, 512, stream);
    int parity = b & 1;
    const unsigned short* W0s = sW0b + (size_t)b * HALF * HDIM;
    const unsigned short* Whs = sWhb + (size_t)b * NLAY * HDIM * HDIM;
    const unsigned short* Wos = sWob + (size_t)b * HDIM * HALF;
    const unsigned short* W0t = tW0b + (size_t)b * HALF * HDIM;
    const unsigned short* Wht = tWhb + (size_t)b * NLAY * HDIM * HDIM;
    const unsigned short* Wot = tWob + (size_t)b * HDIM * HALF;

    for (int ch = 0; ch < BTOT / CHUNK; ++ch) {
      int rowBase = ch * CHUNK;
      const float* Achunk = yCur + (size_t)rowBase * DDIM;
      // --- scale net (tanh) ---
      gemm(Achunk, 1, DDIM, parity ? HALF : 0, W0s, sb0 + b * HDIM,
           actS0, 0, CHUNK, HDIM, HALF);
      gemm(actS0, 0, 0, 0, Whs + 0 * (size_t)HDIM * HDIM,
           sbh + (b * NLAY + 0) * HDIM, actS1, 0, CHUNK, HDIM, HDIM);
      gemm(actS1, 0, 0, 0, Whs + 1 * (size_t)HDIM * HDIM,
           sbh + (b * NLAY + 1) * HDIM, actS0, 0, CHUNK, HDIM, HDIM);
      gemm(actS0, 0, 0, 0, Whs + 2 * (size_t)HDIM * HDIM,
           sbh + (b * NLAY + 2) * HDIM, actS1, 0, CHUNK, HDIM, HDIM);
      gemm(actS1, 0, 0, 0, Wos, sbo + b * HALF,
           sout, 2, CHUNK, HALF, HDIM);
      // --- translate net (relu) ---
      gemm(Achunk, 1, DDIM, parity ? HALF : 0, W0t, tb0 + b * HDIM,
           actT0, 1, CHUNK, HDIM, HALF);
      gemm(actT0, 0, 0, 0, Wht + 0 * (size_t)HDIM * HDIM,
           tbh + (b * NLAY + 0) * HDIM, actT1, 1, CHUNK, HDIM, HDIM);
      gemm(actT1, 0, 0, 0, Wht + 1 * (size_t)HDIM * HDIM,
           tbh + (b * NLAY + 1) * HDIM, actT0, 1, CHUNK, HDIM, HDIM);
      gemm(actT0, 0, 0, 0, Wht + 2 * (size_t)HDIM * HDIM,
           tbh + (b * NLAY + 2) * HDIM, actT1, 1, CHUNK, HDIM, HDIM);
      gemm(actT1, 0, 0, 0, Wot, tbo + b * HALF,
           tout, 2, CHUNK, HALF, HDIM);

      nf_couple<<<CHUNK / 8, 256, 0, stream>>>(sout, tout, yCur, z, logdet,
                                               stats, rowBase, parity);
    }
    nf_bnstats<<<1, 64, 0, stream>>>(stats, lg + b * DDIM, beta + b * DDIM,
                                     bnp, BTOT);
    nf_bnapply<<<(BTOT * DDIM) / 256, 256, 0, stream>>>(z, bnp, yCur, logdet,
                                                        (long)BTOT * DDIM);
  }

  (void)hipMemcpyAsync(d_out, yCur, (size_t)BTOT * DDIM * 4,
                       hipMemcpyDeviceToDevice, stream);
  (void)hipMemcpyAsync((float*)d_out + (size_t)BTOT * DDIM, logdet,
                       (size_t)BTOT * 4, hipMemcpyDeviceToDevice, stream);
}
